// GeometricAttention_18339510354287
// MI455X (gfx1250) — matrix operand order repaired and checked by reading, not compiled, not run on hardware
//
#include <hip/hip_runtime.h>
#include <hip/hip_bf16.h>

#define EMBED_DIM   1024
#define NUM_NEURONS 256
#define ROWS_PER_WG 64
#define THREADS     128
#define KC          32

typedef __bf16        bf16x16 __attribute__((ext_vector_type(16)));
typedef float         v8f     __attribute__((ext_vector_type(8)));
typedef unsigned int  u32x4   __attribute__((ext_vector_type(4)));

union Frag { bf16x16 v; u32x4 u4[2]; };

__device__ __forceinline__ unsigned short f2bf(float f) {
  unsigned u = __float_as_uint(f);
  u += 0x7FFFu + ((u >> 16) & 1u);   // round-to-nearest-even
  return (unsigned short)(u >> 16);
}

__global__ __launch_bounds__(THREADS)
void geo_attn_kernel(const float* __restrict__ x,
                     const float* __restrict__ positions,
                     const float* __restrict__ values,
                     const float* __restrict__ temperature,
                     float* __restrict__ out) {
  // Phase-1 staging (X row-major, P transposed K-major), bf16
  __shared__ __align__(16) unsigned short sX [ROWS_PER_WG * KC];          //  4 KB
  __shared__ __align__(16) unsigned short sPT[KC * NUM_NEURONS];          // 16 KB
  // Phase-3: normalized weights (bf16) + V chunk
  __shared__ __align__(16) unsigned short sW [ROWS_PER_WG * NUM_NEURONS]; // 32 KB
  __shared__ __align__(16) unsigned short sV [KC * 128];                  //  8 KB
  __shared__ float x2s [ROWS_PER_WG];
  __shared__ float p2s [NUM_NEURONS];
  __shared__ float invT[NUM_NEURONS];

  const int tid    = threadIdx.x;
  const int lane   = tid & 31;
  const int waveId = tid >> 5;
  const int h      = lane >> 4;          // lane half (K-half for A, K-group for B)
  const int l16    = lane & 15;          // M index for A, N index for C/D
  const int m0     = waveId * 16;        // this wave's row block
  const int krow   = l16 + 16 * h;       // unused; B lanes are indexed by column n = l16
  const long long r0 = (long long)blockIdx.x * ROWS_PER_WG;

  // ---------------- Phase 0: norms & temperature ----------------
  for (int n = tid; n < NUM_NEURONS; n += THREADS) {
    invT[n] = 1.0f / (fabsf(temperature[n]) + 0.1f);
    const float4* pr = (const float4*)(positions + (size_t)n * EMBED_DIM);
    float s = 0.f;
    for (int i = 0; i < EMBED_DIM / 4; ++i) {
      float4 v = pr[i];
      s += v.x * v.x + v.y * v.y + v.z * v.z + v.w * v.w;
    }
    p2s[n] = s;
  }
  if (tid < ROWS_PER_WG) {
    const float4* xr = (const float4*)(x + (size_t)(r0 + tid) * EMBED_DIM);
    float s = 0.f;
    for (int i = 0; i < EMBED_DIM / 4; ++i) {
      float4 v = xr[i];
      s += v.x * v.x + v.y * v.y + v.z * v.z + v.w * v.w;
    }
    x2s[tid] = s;
  }
  __syncthreads();

  // ---------------- Phase 1: S = X * P^T via WMMA bf16 ----------------
  v8f zero;
#pragma unroll
  for (int i = 0; i < 8; ++i) zero[i] = 0.f;

  v8f acc[16];                     // this wave's 16x256 score strip (f32)
#pragma unroll
  for (int i = 0; i < 16; ++i) acc[i] = zero;

  for (int kc = 0; kc < EMBED_DIM; kc += KC) {
#pragma unroll
    for (int i = 0; i < (ROWS_PER_WG * KC) / THREADS; ++i) {  // 16 iters
      int idx = tid + i * THREADS;
      int row = idx >> 5, k = idx & 31;
      sX[idx] = f2bf(x[(size_t)(r0 + row) * EMBED_DIM + kc + k]);
    }
#pragma unroll 8
    for (int i = 0; i < (KC * NUM_NEURONS) / THREADS; ++i) {  // 64 iters
      int idx = tid + i * THREADS;
      int n = idx >> 5, k = idx & 31;
      sPT[n * KC + k] = f2bf(positions[(size_t)n * EMBED_DIM + kc + k]);
    }
    __syncthreads();

    Frag A;
    const u32x4* pa = (const u32x4*)(sX + (m0 + l16) * KC);
    A.u4[0] = pa[h];          // K = h*8 .. h*8+7
    A.u4[1] = pa[2 + h];      // K = 16 + h*8 ..
#pragma unroll
    for (int nt = 0; nt < 16; ++nt) {
      Frag B;
      const u32x4* pb = (const u32x4*)(sPT + (nt * 16 + l16) * KC);
      B.u4[0] = pb[h];
      B.u4[1] = pb[2 + h];
      acc[nt] = __builtin_amdgcn_wmma_f32_16x16x32_bf16(
          false, A.v, false, B.v, (short)0, acc[nt], false, false);
    }
    __syncthreads();
  }

  // ---------------- Phase 2: dist -> exp -> row-normalize ----------------
  float part[8];
#pragma unroll
  for (int r = 0; r < 8; ++r) part[r] = 0.f;

#pragma unroll
  for (int nt = 0; nt < 16; ++nt) {
    int n = nt * 16 + l16;
    float p2v = p2s[n];
    float it  = invT[n];
#pragma unroll
    for (int r = 0; r < 8; ++r) {
      int m = m0 + r + 8 * h;                 // C layout: VGPR r, lane half h
      float d2 = x2s[m] + p2v - 2.0f * acc[nt][r];
      float d  = sqrtf(fmaxf(d2, 0.0f));
      float wv = __expf(-d * it);
      acc[nt][r] = wv;
      part[r]   += wv;
    }
  }
#pragma unroll
  for (int r = 0; r < 8; ++r) {               // reduce across the 16-lane half
    float s = part[r];
    s += __shfl_xor(s, 1, 32);
    s += __shfl_xor(s, 2, 32);
    s += __shfl_xor(s, 4, 32);
    s += __shfl_xor(s, 8, 32);
    part[r] = 1.0f / (s + 1e-8f);
  }
#pragma unroll
  for (int nt = 0; nt < 16; ++nt) {
#pragma unroll
    for (int r = 0; r < 8; ++r) {
      sW[(m0 + r + 8 * h) * NUM_NEURONS + nt * 16 + l16] =
          f2bf(acc[nt][r] * part[r]);
    }
  }
  __syncthreads();

  // ---------------- Phase 3: out = W * V via WMMA bf16 ----------------
  for (int dc = 0; dc < EMBED_DIM; dc += 128) {
    v8f oacc[8];
#pragma unroll
    for (int i = 0; i < 8; ++i) oacc[i] = zero;

    for (int kc = 0; kc < NUM_NEURONS; kc += KC) {
      __syncthreads();                       // protect sV from prior readers
#pragma unroll 8
      for (int i = 0; i < (KC * 128) / THREADS; ++i) {  // 32 iters
        int idx = tid + i * THREADS;
        int k = idx >> 7, d = idx & 127;
        sV[d * KC + k] = f2bf(values[(size_t)(kc + k) * EMBED_DIM + dc + d]);
      }
      __syncthreads();

      Frag A;
      const u32x4* pa = (const u32x4*)(sW + (m0 + l16) * NUM_NEURONS + kc);
      A.u4[0] = pa[h];
      A.u4[1] = pa[2 + h];
#pragma unroll
      for (int dt = 0; dt < 8; ++dt) {
        Frag B;
        const u32x4* pb = (const u32x4*)(sV + (dt * 16 + l16) * KC);
        B.u4[0] = pb[h];
        B.u4[1] = pb[2 + h];
        oacc[dt] = __builtin_amdgcn_wmma_f32_16x16x32_bf16(
            false, A.v, false, B.v, (short)0, oacc[dt], false, false);
      }
    }
#pragma unroll
    for (int dt = 0; dt < 8; ++dt) {
#pragma unroll
      for (int r = 0; r < 8; ++r) {
        out[(size_t)(r0 + m0 + r + 8 * h) * EMBED_DIM + dc + dt * 16 + l16] =
            oacc[dt][r];
      }
    }
  }
}

extern "C" void kernel_launch(void* const* d_in, const int* in_sizes, int n_in,
                              void* d_out, int out_size, void* d_ws, size_t ws_size,
                              hipStream_t stream) {
  const float* x           = (const float*)d_in[0];
  const float* positions   = (const float*)d_in[1];
  const float* values      = (const float*)d_in[2];
  const float* temperature = (const float*)d_in[3];
  float* out = (float*)d_out;

  int BT = in_sizes[0] / EMBED_DIM;      // 8 * 4096 = 32768
  int grid = BT / ROWS_PER_WG;           // 512 workgroups, 4 waves each
  geo_attn_kernel<<<grid, THREADS, 0, stream>>>(x, positions, values, temperature, out);
}
